// NeWCRFBlock_38809324487101
// MI455X (gfx1250) — matrix operand order repaired and checked by reading, not compiled, not run on hardware
//
#include <hip/hip_runtime.h>
#include <hip/hip_bf16.h>

// ---------- types ----------
typedef __attribute__((ext_vector_type(16))) _Float16 v16h;
typedef __attribute__((ext_vector_type(8)))  _Float16 v8h;
typedef __attribute__((ext_vector_type(4)))  _Float16 v4h;
typedef __attribute__((ext_vector_type(8)))  float    v8f;

#define BM 128
#define BN 128
#define BK 32
#define AST 40   // BK + 8 halfs pad (80B rows -> 16B aligned)
#define PST 72   // 64 + 8 halfs pad (144B rows -> 16B aligned)

// EPI variants
#define EPI_GELU_H16 1   // bias + gelu -> f16 [M,N]
#define EPI_WO       2   // bias + residual (rowmapped natural layout) -> f32
#define EPI_ADD_OUT  3   // bias + residual -> f32 (final output)
#define EPI_QKV      4   // bias -> f16 in (win,head,tok,hd) layout
#define EPI_F32      5   // bias -> f32 [M,N]

// ---------- CDNA5 async global->LDS path (guarded; falls back to sync copy) ----------
#if __has_builtin(__builtin_amdgcn_global_load_async_to_lds_b128)
#define HAVE_ASYNC_LDS 1
#else
#define HAVE_ASYNC_LDS 0
#endif

// builtin signature (from hipcc diagnostic): params are int4-vector pointers
typedef int i4v __attribute__((vector_size(16)));
typedef __attribute__((address_space(1))) i4v* gp_i4;
typedef __attribute__((address_space(3))) i4v* lp_i4;

#if HAVE_ASYNC_LDS
#if __has_builtin(__builtin_amdgcn_s_wait_asynccnt)
#define WAIT_ASYNC(n) __builtin_amdgcn_s_wait_asynccnt(n)
#else
#define WAIT_ASYNC(n) asm volatile("s_wait_asynccnt %0" ::"n"(n) : "memory")
#endif
#else
#define WAIT_ASYNC(n) do { } while (0)
#endif

// copy 32 bytes (16 halfs) global -> LDS
__device__ __forceinline__ void cp32(const _Float16* g, _Float16* l) {
#if HAVE_ASYNC_LDS
  __builtin_amdgcn_global_load_async_to_lds_b128((gp_i4)(g),     (lp_i4)(l),     0, 0);
  __builtin_amdgcn_global_load_async_to_lds_b128((gp_i4)(g + 8), (lp_i4)(l + 8), 0, 0);
#else
  *(v8h*)l       = *(const v8h*)g;
  *(v8h*)(l + 8) = *(const v8h*)(g + 8);
#endif
}

__device__ __forceinline__ float gelu_f(float x) {
  return 0.5f * x * (1.0f + erff(x * 0.70710678118654752f));
}

// window-row -> natural-row map (B=8, H=W=128, ws=8 hardcoded)
__device__ __forceinline__ int win_to_nat(int gr) {
  int w = gr >> 6, n = gr & 63;
  int b = w >> 8, hb = (w >> 4) & 15, wb = w & 15;
  int r = n >> 3, c = n & 7;
  return (b << 14) + ((hb * 8 + r) << 7) + (wb * 8 + c);
}

// A-matrix 16x32 f16 fragment (ISA 7.12.2): row = lane&15,
// elems 0..7 -> K = (lane>>4)*8 + e ; elems 8..15 -> K = 16 + (lane>>4)*8 + e
__device__ __forceinline__ v16h frag_a(const _Float16* base, int stride, int lane) {
  const _Float16* p = base + (lane & 15) * stride + ((lane >> 4) << 3);
  v8h lo = *(const v8h*)p;
  v8h hi = *(const v8h*)(p + 16);
  v16h r;
#pragma unroll
  for (int i = 0; i < 8; ++i) { r[i] = lo[i]; r[i + 8] = hi[i]; }
  return r;
}

// B-matrix 32x16 f16 fragment from B^T rows in LDS:
// elems 0..7 -> K = (lane>>4)*8 + e ; elems 8..15 -> K = 16 + (lane>>4)*8 + e
__device__ __forceinline__ v16h frag_b(const _Float16* base, int stride, int lane) {
  const _Float16* p = base + (lane & 15) * stride + ((lane >> 4) << 3);
  v8h lo = *(const v8h*)p;
  v8h hi = *(const v8h*)(p + 16);
  v16h r;
#pragma unroll
  for (int i = 0; i < 8; ++i) { r[i] = lo[i]; r[i + 8] = hi[i]; }
  return r;
}

// ---------- prep: f32 -> f16 convert (4 elems/thread) ----------
__global__ __launch_bounds__(256)
void cvt_f16_kernel(const float* __restrict__ src, _Float16* __restrict__ dst) {
  int i = blockIdx.x * 256 + threadIdx.x;
  float4 f = ((const float4*)src)[i];
  v4h h;
  h[0] = (_Float16)f.x; h[1] = (_Float16)f.y; h[2] = (_Float16)f.z; h[3] = (_Float16)f.w;
  ((v4h*)dst)[i] = h;
}

// ---------- prep: W (K x N f32) -> WT (N x K f16) ----------
__global__ __launch_bounds__(256)
void transpose_cvt_kernel(const float* __restrict__ W, _Float16* __restrict__ WT,
                          int K, int N) {
  __shared__ float t[32][33];
  const int tx = threadIdx.x & 31, ty = threadIdx.x >> 5;  // ty: 0..7
  const int n0 = blockIdx.x << 5, k0 = blockIdx.y << 5;
#pragma unroll
  for (int i = 0; i < 4; ++i) {
    int k = (ty << 2) + i;
    t[k][tx] = W[(size_t)(k0 + k) * N + n0 + tx];
  }
  __syncthreads();
#pragma unroll
  for (int i = 0; i < 4; ++i) {
    int n = (ty << 2) + i;
    WT[(size_t)(n0 + n) * K + k0 + tx] = (_Float16)t[tx][n];
  }
}

// ---------- tiled GEMM: out = epilogue(A @ W + bias), W given as WT (N x K f16) ----------
// 128x128 block, 8 waves, each wave owns a 32x64 tile (8 WMMAs / K-step),
// double-buffered LDS fed by async global->LDS copies.
template <int EPI>
__global__ __launch_bounds__(256, 2)
void gemm_kernel(const _Float16* __restrict__ A, const _Float16* __restrict__ WT,
                 const float* __restrict__ bias, void* __restrict__ Out,
                 const float* __restrict__ Resid, int M, int K, int N) {
  __shared__ __align__(16) _Float16 As[2][BM * AST];
  __shared__ __align__(16) _Float16 Bs[2][BN * AST];
  const int tid  = threadIdx.x;
  const int wave = tid >> 5, lane = tid & 31;
  const int wm = wave & 3, wn = wave >> 2;      // wave tile: rows wm*32, cols wn*64
  const int row0 = blockIdx.x * BM;
  const int col0 = blockIdx.y * BN;

  v8f acc[2][4];
#pragma unroll
  for (int mi = 0; mi < 2; ++mi)
#pragma unroll
    for (int j = 0; j < 4; ++j)
#pragma unroll
      for (int i = 0; i < 8; ++i) acc[mi][j][i] = 0.0f;

  const int ar   = tid >> 1;          // 2 threads per LDS row (128 rows)
  const int aseg = (tid & 1) << 4;    // 16 halfs each
  const _Float16* Athr = A  + (size_t)(row0 + ar) * K + aseg;
  const _Float16* Bthr = WT + (size_t)(col0 + ar) * K + aseg;
  _Float16* Adst0 = &As[0][ar * AST + aseg];
  _Float16* Adst1 = &As[1][ar * AST + aseg];
  _Float16* Bdst0 = &Bs[0][ar * AST + aseg];
  _Float16* Bdst1 = &Bs[1][ar * AST + aseg];

  const int nsteps = K / BK;
  // prologue: stage tile 0 into buffer 0
  cp32(Athr, Adst0);
  cp32(Bthr, Bdst0);

  for (int t = 0; t < nsteps; ++t) {
    const int cur = t & 1;
    if (t + 1 < nsteps) {
      const int koff = (t + 1) * BK;
      cp32(Athr + koff, cur ? Adst0 : Adst1);
      cp32(Bthr + koff, cur ? Bdst0 : Bdst1);
      WAIT_ASYNC(4);   // tile t done; tile t+1 still in flight
    } else {
      WAIT_ASYNC(0);
    }
    __syncthreads();

    const _Float16* Ab = &As[cur][(wm << 5) * AST];
    const _Float16* Bb = &Bs[cur][(wn << 6) * AST];
    v16h a0 = frag_a(Ab, AST, lane);
    v16h a1 = frag_a(Ab + 16 * AST, AST, lane);
#pragma unroll
    for (int j = 0; j < 4; ++j) {
      v16h b = frag_b(Bb + (j << 4) * AST, AST, lane);
      acc[0][j] = __builtin_amdgcn_wmma_f32_16x16x32_f16(false, a0, false, b,
                                                         (short)0, acc[0][j], false, false);
      acc[1][j] = __builtin_amdgcn_wmma_f32_16x16x32_f16(false, a1, false, b,
                                                         (short)0, acc[1][j], false, false);
    }
    __syncthreads();
  }

  // ---- epilogue ----
  const int cin = lane & 15;
#pragma unroll
  for (int mi = 0; mi < 2; ++mi) {
    const int rbase = row0 + (wm << 5) + (mi << 4) + ((lane >> 4) << 3);
#pragma unroll
    for (int j = 0; j < 4; ++j) {
      const int gc = col0 + (wn << 6) + (j << 4) + cin;
      const float bv = bias[gc];
#pragma unroll
      for (int i = 0; i < 8; ++i) {
        const int gr = rbase + i;
        float val = acc[mi][j][i] + bv;
        if (EPI == EPI_GELU_H16) {
          ((_Float16*)Out)[(size_t)gr * N + gc] = (_Float16)gelu_f(val);
        } else if (EPI == EPI_F32) {
          ((float*)Out)[(size_t)gr * N + gc] = val;
        } else if (EPI == EPI_QKV) {
          int w = gr >> 6, n = gr & 63, hh = gc >> 5, d = gc & 31;
          ((_Float16*)Out)[(((size_t)(w * 8 + hh)) << 11) + (n << 5) + d] = (_Float16)val;
        } else if (EPI == EPI_WO) {
          int m = win_to_nat(gr);
          size_t o = (size_t)m * N + gc;
          ((float*)Out)[o] = val + Resid[o];
        } else if (EPI == EPI_ADD_OUT) {
          size_t o = (size_t)gr * N + gc;
          ((float*)Out)[o] = val + Resid[o];
        }
      }
    }
  }
}

// ---------- window gather + LayerNorm: feature (B,C,H,W) -> xn f16 (win,tok,C) ----------
__global__ __launch_bounds__(256)
void ln_win_kernel(const float* __restrict__ feature, const float* __restrict__ g,
                   const float* __restrict__ beta, _Float16* __restrict__ xn) {
  __shared__ float tile[64][257];
  __shared__ float mus[64], rstds[64];
  const int w = blockIdx.x;                    // 0..2047
  const int b = w >> 8, hb = (w >> 4) & 15, wb = w & 15;
  const int tid = threadIdx.x;
  const int n = tid & 63;
  const int cgrp = tid >> 6;                   // 0..3
  const int r = n >> 3, cc = n & 7;
  const size_t pixoff = (size_t)(hb * 8 + r) * 128 + (wb * 8 + cc);
  const size_t bbase = (size_t)b * 256 * 16384;
  for (int i = 0; i < 64; ++i) {
    int ch = (cgrp << 6) + i;
    tile[n][ch] = feature[bbase + (size_t)ch * 16384 + pixoff];
  }
  __syncthreads();
  const int wave = tid >> 5, lane = tid & 31;
  for (int j = 0; j < 8; ++j) {
    int tok = (wave << 3) + j;
    float s1 = 0.f, s2 = 0.f;
#pragma unroll
    for (int i = 0; i < 8; ++i) {
      float x = tile[tok][lane + (i << 5)];
      s1 += x; s2 += x * x;
    }
#pragma unroll
    for (int m = 16; m >= 1; m >>= 1) {
      s1 += __shfl_xor(s1, m);
      s2 += __shfl_xor(s2, m);
    }
    if (lane == 0) {
      float mu = s1 * (1.0f / 256.0f);
      float var = s2 * (1.0f / 256.0f) - mu * mu;
      mus[tok] = mu;
      rstds[tok] = rsqrtf(var + 1e-5f);
    }
  }
  __syncthreads();
  const size_t obase = ((size_t)w) << 14;      // w*64*256
  for (int i = 0; i < 64; ++i) {
    int idx = tid + (i << 8);
    int tt = idx >> 8, ch = idx & 255;
    float val = (tile[tt][ch] - mus[tt]) * rstds[tt] * g[ch] + beta[ch];
    xn[obase + idx] = (_Float16)val;
  }
}

// ---------- row LayerNorm: x2 f32 (M,256) -> f16 ----------
__global__ __launch_bounds__(256)
void ln_rows_kernel(const float* __restrict__ X, const float* __restrict__ g,
                    const float* __restrict__ beta, _Float16* __restrict__ out) {
  const int wave = threadIdx.x >> 5, lane = threadIdx.x & 31;
  const int row = blockIdx.x * 8 + wave;
  const float* xr = X + (size_t)row * 256;
  float v[8];
  float s1 = 0.f, s2 = 0.f;
#pragma unroll
  for (int i = 0; i < 8; ++i) {
    v[i] = xr[lane + (i << 5)];
    s1 += v[i]; s2 += v[i] * v[i];
  }
#pragma unroll
  for (int m = 16; m >= 1; m >>= 1) {
    s1 += __shfl_xor(s1, m);
    s2 += __shfl_xor(s2, m);
  }
  float mu = s1 * (1.0f / 256.0f);
  float var = s2 * (1.0f / 256.0f) - mu * mu;
  float rstd = rsqrtf(var + 1e-5f);
#pragma unroll
  for (int i = 0; i < 8; ++i) {
    int ch = lane + (i << 5);
    out[(size_t)row * 256 + ch] = (_Float16)((v[i] - mu) * rstd * g[ch] + beta[ch]);
  }
}

// ---------- window attention: one block per (window, head) ----------
__global__ __launch_bounds__(128)
void attn_kernel(const _Float16* __restrict__ q, const _Float16* __restrict__ k,
                 const _Float16* __restrict__ v, const float* __restrict__ rpb,
                 const int* __restrict__ rel_idx, _Float16* __restrict__ o) {
  __shared__ __align__(16) _Float16 qs[64 * AST];
  __shared__ __align__(16) _Float16 ks[64 * AST];
  __shared__ __align__(16) _Float16 vts[32 * PST];
  __shared__ __align__(16) _Float16 ps[64 * PST];
  __shared__ float sc[64][65];

  const int wh = blockIdx.x;                   // 0..16383
  const int w = wh >> 3, h = wh & 7;
  const int tid = threadIdx.x;
  const int wave = tid >> 5, lane = tid & 31;
  const size_t base = (size_t)wh << 11;        // * 2048 elements

  // stage q,k via async copies (row-major 64x32)
  {
    int rr = tid >> 1, seg = (tid & 1) << 4;
    cp32(q + base + (rr << 5) + seg, &qs[rr * AST + seg]);
    cp32(k + base + (rr << 5) + seg, &ks[rr * AST + seg]);
  }
  // stage v transposed: vts[d][m]
#pragma unroll
  for (int i = 0; i < 16; ++i) {
    int idx = tid + (i << 7);
    int m = idx >> 5, d = idx & 31;
    vts[d * PST + m] = v[base + idx];
  }
  WAIT_ASYNC(0);
  __syncthreads();

  // scores = q @ k^T  (each wave: 16-row strip x 64 cols)
  const float scale = 0.17677669529663687f;    // 1/sqrt(32)
  v16h qa = frag_a(&qs[(wave << 4) * AST], AST, lane);
  const int rb = (wave << 4) + ((lane >> 4) << 3);
  const int cin = lane & 15;
#pragma unroll
  for (int j = 0; j < 4; ++j) {
    v8f s;
#pragma unroll
    for (int i = 0; i < 8; ++i) s[i] = 0.0f;
    v16h kf = frag_b(&ks[(j << 4) * AST], AST, lane);
    s = __builtin_amdgcn_wmma_f32_16x16x32_f16(false, qa, false, kf, (short)0, s, false, false);
#pragma unroll
    for (int i = 0; i < 8; ++i) {
      int row = rb + i, col = (j << 4) + cin;
      float bv = rpb[rel_idx[(row << 6) + col] * 8 + h];
      sc[row][col] = s[i] * scale + bv;
    }
  }
  __syncthreads();

  // softmax per row (threads 0..63 each own one row)
  if (tid < 64) {
    float mx = -1e30f;
    for (int m2 = 0; m2 < 64; ++m2) mx = fmaxf(mx, sc[tid][m2]);
    float sum = 0.f;
    for (int m2 = 0; m2 < 64; ++m2) sum += __expf(sc[tid][m2] - mx);
    float inv = 1.0f / sum;
    for (int m2 = 0; m2 < 64; ++m2)
      ps[tid * PST + m2] = (_Float16)(__expf(sc[tid][m2] - mx) * inv);
  }
  __syncthreads();

  // o = probs @ v  (K = 64 -> two WMMA steps; N = 32 -> two 16-col tiles)
  v8f oacc[2];
#pragma unroll
  for (int j2 = 0; j2 < 2; ++j2)
#pragma unroll
    for (int i = 0; i < 8; ++i) oacc[j2][i] = 0.0f;
#pragma unroll
  for (int ks2 = 0; ks2 < 64; ks2 += 32) {
    v16h pa = frag_a(&ps[(wave << 4) * PST + ks2], PST, lane);
#pragma unroll
    for (int j2 = 0; j2 < 2; ++j2) {
      v16h vf = frag_b(&vts[(j2 << 4) * PST + ks2], PST, lane);
      oacc[j2] = __builtin_amdgcn_wmma_f32_16x16x32_f16(false, pa, false, vf,
                                                        (short)0, oacc[j2], false, false);
    }
  }
  // store o in (win, tok, C) layout (col = h*32 + d)
#pragma unroll
  for (int j2 = 0; j2 < 2; ++j2)
#pragma unroll
    for (int i = 0; i < 8; ++i) {
      int row = rb + i, d = (j2 << 4) + cin;
      o[(((size_t)w << 6) + row) * 256 + (h << 5) + d] = (_Float16)oacc[j2][i];
    }
}

// ---------- host launch ----------
extern "C" void kernel_launch(void* const* d_in, const int* in_sizes, int n_in,
                              void* d_out, int out_size, void* d_ws, size_t ws_size,
                              hipStream_t stream) {
  (void)in_sizes; (void)n_in; (void)out_size; (void)ws_size;
  const float* x       = (const float*)d_in[0];
  const float* feature = (const float*)d_in[1];
  const float* Wq = (const float*)d_in[2];
  const float* bq = (const float*)d_in[3];
  const float* Wk = (const float*)d_in[4];
  const float* bk = (const float*)d_in[5];
  const float* Wv = (const float*)d_in[6];
  const float* bv = (const float*)d_in[7];
  const float* Wo = (const float*)d_in[8];
  const float* bo = (const float*)d_in[9];
  const float* rpb = (const float*)d_in[10];
  const float* g1 = (const float*)d_in[11];
  const float* beta1 = (const float*)d_in[12];
  const float* g2 = (const float*)d_in[13];
  const float* beta2 = (const float*)d_in[14];
  const float* Wm1 = (const float*)d_in[15];
  const float* bm1 = (const float*)d_in[16];
  const float* Wm2 = (const float*)d_in[17];
  const float* bm2 = (const float*)d_in[18];
  const float* Wu1 = (const float*)d_in[19];
  const float* bu1 = (const float*)d_in[20];
  const float* Wu2 = (const float*)d_in[21];
  const float* bu2 = (const float*)d_in[22];
  const int* rel_idx = (const int*)d_in[23];

  const int M = 8 * 128 * 128;   // 131072 rows
  const int C = 256, HID = 1024;

  // workspace layout
  char* ws = (char*)d_ws;
  float*    unary = (float*)ws;                                   // M*C f32
  float*    x2    = (float*)(ws + (size_t)M * C * 4);             // M*C f32
  _Float16* qb    = (_Float16*)(ws + (size_t)M * C * 8);          // M*C f16
  _Float16* kb    = qb + (size_t)M * C;
  _Float16* vb    = kb + (size_t)M * C;
  _Float16* ob    = vb + (size_t)M * C;
  _Float16* h1    = qb;                                           // M*HID f16 (reuse q..o)
  _Float16* xn    = (_Float16*)(ws + (size_t)M * C * 8 + (size_t)M * HID * 2);
  _Float16* u1    = xn + (size_t)M * C;
  _Float16* x16   = u1 + (size_t)M * C;                           // f16 copy of x
  _Float16* WTq   = x16 + (size_t)M * C;                          // transposed f16 weights
  _Float16* WTk   = WTq + C * C;
  _Float16* WTv   = WTk + C * C;
  _Float16* WTo   = WTv + C * C;
  _Float16* WTu1  = WTo + C * C;
  _Float16* WTu2  = WTu1 + C * C;
  _Float16* WTm1  = WTu2 + C * C;                                 // HID x C
  _Float16* WTm2  = WTm1 + (size_t)HID * C;                       // C x HID

  dim3 blk(256);
  dim3 gC(M / BM, C / BN);       // (1024, 2)
  dim3 gH(M / BM, HID / BN);     // (1024, 8)
  dim3 tCC(C / 32, C / 32);      // (8, 8)
  dim3 tM1(HID / 32, C / 32);    // W (C x HID) -> WT (HID x C)
  dim3 tM2(C / 32, HID / 32);    // W (HID x C) -> WT (C x HID)

  // ---- prep: f16 conversions / weight transposes ----
  cvt_f16_kernel<<<dim3((M * C) / 4 / 256), blk, 0, stream>>>(x, x16);
  transpose_cvt_kernel<<<tCC, blk, 0, stream>>>(Wq,  WTq,  C, C);
  transpose_cvt_kernel<<<tCC, blk, 0, stream>>>(Wk,  WTk,  C, C);
  transpose_cvt_kernel<<<tCC, blk, 0, stream>>>(Wv,  WTv,  C, C);
  transpose_cvt_kernel<<<tCC, blk, 0, stream>>>(Wo,  WTo,  C, C);
  transpose_cvt_kernel<<<tCC, blk, 0, stream>>>(Wu1, WTu1, C, C);
  transpose_cvt_kernel<<<tCC, blk, 0, stream>>>(Wu2, WTu2, C, C);
  transpose_cvt_kernel<<<tM1, blk, 0, stream>>>(Wm1, WTm1, C, HID);
  transpose_cvt_kernel<<<tM2, blk, 0, stream>>>(Wm2, WTm2, HID, C);

  // ---- unary path: u1 = gelu(x@Wu1+bu1); unary = u1@Wu2+bu2 ----
  gemm_kernel<EPI_GELU_H16><<<gC, blk, 0, stream>>>(x16, WTu1, bu1, u1,    nullptr, M, C, C);
  gemm_kernel<EPI_F32    ><<<gC, blk, 0, stream>>>(u1,  WTu2, bu2, unary, nullptr, M, C, C);

  // ---- window gather + LN1 ----
  ln_win_kernel<<<dim3(2048), blk, 0, stream>>>(feature, g1, beta1, xn);

  // ---- q,k,v projections into (win,head,tok,hd) f16 ----
  gemm_kernel<EPI_QKV><<<gC, blk, 0, stream>>>(xn, WTq, bq, qb, nullptr, M, C, C);
  gemm_kernel<EPI_QKV><<<gC, blk, 0, stream>>>(xn, WTk, bk, kb, nullptr, M, C, C);
  gemm_kernel<EPI_QKV><<<gC, blk, 0, stream>>>(xn, WTv, bv, vb, nullptr, M, C, C);

  // ---- window attention ----
  attn_kernel<<<dim3(2048 * 8), dim3(128), 0, stream>>>(qb, kb, vb, rpb, rel_idx, ob);

  // ---- x2 = window_reverse(o @ Wo + bo) + unary ----
  gemm_kernel<EPI_WO><<<gC, blk, 0, stream>>>(ob, WTo, bo, x2, unary, M, C, C);

  // ---- LN2 then MLP: out = x2 + gelu(ln(x2)@Wm1+bm1)@Wm2+bm2 ----
  ln_rows_kernel<<<dim3(M / 8), blk, 0, stream>>>(x2, g2, beta2, xn);
  gemm_kernel<EPI_GELU_H16><<<gH, blk, 0, stream>>>(xn, WTm1, bm1, h1, nullptr, M, C, HID);
  gemm_kernel<EPI_ADD_OUT ><<<gC, blk, 0, stream>>>(h1, WTm2, bm2, (float*)d_out, x2, M, HID, C);
}
